// BiRecurrentLayer_52037823758735
// MI455X (gfx1250) — matrix operand order repaired and checked by reading, not compiled, not run on hardware
//
#include <hip/hip_runtime.h>
#include <hip/hip_bf16.h>

#define B_   32
#define T_   512
#define D_   512
#define H_   512
#define G4_  2048
#define KW_  1024           // rows of W (D+H)
#define M_   (B_ * T_)      // 16384 rows of xp
#define NWG_ 16             // workgroups per direction in the scan kernel

typedef __bf16 bf16_t;
typedef bf16_t v16bf __attribute__((ext_vector_type(16)));
typedef float  v8f   __attribute__((ext_vector_type(8)));
typedef int    v4i   __attribute__((ext_vector_type(4)));
typedef unsigned short u16;

union FragBF {
    v16bf v;
    uint4 q[2];
    u16   us[16];
};

__device__ __forceinline__ u16 f2bf(float f) {
    unsigned u = __float_as_uint(f);
    u += 0x7FFFu + ((u >> 16) & 1u);   // round-to-nearest-even
    return (u16)(u >> 16);
}
__device__ __forceinline__ float sigf(float x) { return 1.0f / (1.0f + __expf(-x)); }

// ---- CDNA5 async global->LDS copy (ASYNCcnt-tracked), with guarded fallback
#if defined(__has_builtin)
#if __has_builtin(__builtin_amdgcn_global_load_async_to_lds_b128)
#define HAS_ASYNC_LDS 1
#endif
#endif
#ifndef HAS_ASYNC_LDS
#define HAS_ASYNC_LDS 0
#endif

__device__ __forceinline__ void async_copy_b128(const void* gsrc, void* ldst) {
#if HAS_ASYNC_LDS
    __builtin_amdgcn_global_load_async_to_lds_b128(
        (v4i*)gsrc, (v4i*)ldst, 0, 0);
#else
    *(uint4*)ldst = *(const uint4*)gsrc;
#endif
}
__device__ __forceinline__ void async_wait_4() {
#if HAS_ASYNC_LDS
#if __has_builtin(__builtin_amdgcn_s_wait_asynccnt)
    __builtin_amdgcn_s_wait_asynccnt(4);
#endif
#endif
}
__device__ __forceinline__ void async_wait_0() {
#if HAS_ASYNC_LDS
#if __has_builtin(__builtin_amdgcn_s_wait_asynccnt)
    __builtin_amdgcn_s_wait_asynccnt(0);
#endif
#endif
}

// ---------------------------------------------------------------------------
// Kernel 0: one-time conversions (all L2-resident afterwards):
//   xbf[bt][d]          = bf16(x)                       (16 MB, row-major)
//   wT[dir][n][k]       = bf16(W[dir][k][n])            (8 MB, n-major = exact
//                         per-lane WMMA B-fragment layout for k-contiguous reads)
//   cnt[]               = 0                              (scan step barrier)
// ---------------------------------------------------------------------------
__global__ __launch_bounds__(256) void prep_kernel(
    const float* __restrict__ x,
    const float* __restrict__ W_fw, const float* __restrict__ W_bw,
    u16* __restrict__ xbf, u16* __restrict__ wT, int* __restrict__ cnt)
{
    const int idx = blockIdx.x * 256 + threadIdx.x;
    if (idx < 2 * T_) cnt[idx] = 0;
    const int NE_X = B_ * T_ * D_;                 // 8M
    if (idx < NE_X) {
        xbf[idx] = f2bf(x[idx]);
    } else {
        const int j   = idx - NE_X;                // < 2 * 2048 * 1024 = 4M
        const int k   = j & (KW_ - 1);
        const int n   = (j >> 10) & (G4_ - 1);
        const int dir = j >> 21;
        const float* __restrict__ W = dir ? W_bw : W_fw;
        wT[j] = f2bf(W[(size_t)k * G4_ + n]);
    }
}

// ---------------------------------------------------------------------------
// Kernel 1: xp[dir] = x @ Wx[dir] + b[dir]   (M=16384, K=512, N=2048)
// 128x128 block tile, K-tiles of 32, double-buffered LDS filled by
// GLOBAL_LOAD_ASYNC_TO_LDS_B128 (pipelined one K-tile ahead, s_wait_asynccnt),
// bf16 WMMA with fp32 accumulate. 8 waves: 4x2 grid, 2x4 tiles each.
// ---------------------------------------------------------------------------
__global__ __launch_bounds__(256) void xp_gemm_kernel(
    const u16* __restrict__ xbf, const u16* __restrict__ wT,
    const float* __restrict__ b_fw, const float* __restrict__ b_bw,
    float* __restrict__ xp_ws)
{
    const int dir = blockIdx.z;
    const float* __restrict__ bias = dir ? b_bw : b_fw;
    const u16* __restrict__ wTd = wT + (size_t)dir * G4_ * KW_;
    float* __restrict__ xp = xp_ws + (size_t)dir * M_ * G4_;

    const int n0   = blockIdx.x * 128;
    const int bt0  = blockIdx.y * 128;
    const int tid  = threadIdx.x;
    const int lane = tid & 31;
    const int w    = tid >> 5;
    const int m0c  = (w & 3) * 32;
    const int n0c  = (w >> 2) * 64;
    const int l16  = lane & 15;
    const int lhi  = lane >> 4;

    __shared__ u16 Alds[2][128][40];   // 128x32 bf16 A tile, padded stride
    __shared__ u16 Blds[2][128][40];   // 128(n) x 32(k) bf16 B tile (n-major)

    v8f acc[2][4];
    #pragma unroll
    for (int tn = 0; tn < 4; ++tn) {
        const float bv = bias[n0 + n0c + tn * 16 + l16];
        #pragma unroll
        for (int v = 0; v < 8; ++v) { acc[0][tn][v] = bv; acc[1][tn][v] = bv; }
    }

    // issue one 128x32 A tile + 128x32 B tile as 16B async copies (4/thread)
    auto issue_tile = [&](int k0, int p) {
        #pragma unroll
        for (int i = 0; i < 2; ++i) {
            const int e = tid + i * 256;       // 512 chunks of 16B per matrix
            const int r = e >> 2, q = e & 3;
            async_copy_b128(xbf + (size_t)(bt0 + r) * D_ + k0 + q * 8,
                            &Alds[p][r][q * 8]);
            async_copy_b128(wTd + (size_t)(n0 + r) * KW_ + k0 + q * 8,
                            &Blds[p][r][q * 8]);
        }
    };

    issue_tile(0, 0);
    for (int k0 = 0; k0 < D_; k0 += 32) {
        const int p = (k0 >> 5) & 1;
        if (k0 + 32 < D_) {
            issue_tile(k0 + 32, p ^ 1);   // prefetch next tile under compute
            async_wait_4();               // drain only the older tile's copies
        } else {
            async_wait_0();
        }
        __syncthreads();

        FragBF af[2];
        #pragma unroll
        for (int tm = 0; tm < 2; ++tm) {
            const u16* pa = &Alds[p][m0c + tm * 16 + l16][lhi * 8];
            af[tm].q[0] = *(const uint4*)pa;
            af[tm].q[1] = *(const uint4*)(pa + 16);
        }
        #pragma unroll
        for (int tn = 0; tn < 4; ++tn) {
            FragBF bf;
            const u16* pb = &Blds[p][n0c + tn * 16 + l16][lhi * 8];
            bf.q[0] = *(const uint4*)pb;
            bf.q[1] = *(const uint4*)(pb + 16);
            #pragma unroll
            for (int tm = 0; tm < 2; ++tm) {
                acc[tm][tn] = __builtin_amdgcn_wmma_f32_16x16x32_bf16(
                    false, af[tm].v, false, bf.v, (short)0, acc[tm][tn], false, false);
            }
        }
        __syncthreads();
    }

    #pragma unroll
    for (int tm = 0; tm < 2; ++tm)
        #pragma unroll
        for (int tn = 0; tn < 4; ++tn)
            #pragma unroll
            for (int v = 0; v < 8; ++v) {
                const int row = bt0 + m0c + tm * 16 + v + lhi * 8;
                const int col = n0 + n0c + tn * 16 + l16;
                xp[(size_t)row * G4_ + col] = acc[tm][tn][v];
            }
}

// ---------------------------------------------------------------------------
// Kernel 2: persistent bidirectional LSTM scan (latency-critical sequential
// part). 16 WGs/direction; WG owns h-columns [wg*32, +32) => all 4 gate
// slices. Wh B-fragments stream straight from L2 (wT rows k=512..1023),
// h double-buffered in global bf16, c-state resident in LDS, per-step
// device-scope acquire/release barrier.
// ---------------------------------------------------------------------------
__global__ __launch_bounds__(256) void lstm_scan_kernel(
    const float* __restrict__ xp_ws, const u16* __restrict__ wT,
    u16* __restrict__ h_ws, int* __restrict__ cnt_ws, float* __restrict__ out)
{
    const int dir = blockIdx.x >> 4;
    const int wg  = blockIdx.x & (NWG_ - 1);
    const float* __restrict__ xp  = xp_ws + (size_t)dir * M_ * G4_;
    const u16*   __restrict__ wTd = wT + (size_t)dir * G4_ * KW_;
    int* __restrict__ cnt = cnt_ws + dir * T_;

    const int tid  = threadIdx.x;
    const int lane = tid & 31;
    const int w    = tid >> 5;
    const int g    = w >> 1;          // gate 0..3 owned by this wave
    const int m0   = (w & 1) * 16;    // batch-row half
    const int l16  = lane & 15;
    const int lhi  = lane >> 4;

    __shared__ float zbuf[4][32][33];
    __shared__ float cbuf[32][32];
    #pragma unroll
    for (int i = 0; i < 4; ++i) ((float*)cbuf)[tid + i * 256] = 0.0f;
    __syncthreads();

    // B-fragment pointers: wT[dir][n = g*512+wg*32+col][k = 512 + ...]
    const u16* bp0 = wTd + (size_t)(g * 512 + wg * 32 + 0  + l16) * KW_ + 512 + lhi * 8;
    const u16* bp1 = wTd + (size_t)(g * 512 + wg * 32 + 16 + l16) * KW_ + 512 + lhi * 8;

    for (int step = 0; step < T_; ++step) {
        const int t = dir ? (T_ - 1 - step) : step;

        if (step > 0) {
            if (tid == 0) {
                while (__hip_atomic_load(&cnt[step - 1], __ATOMIC_ACQUIRE,
                                         __HIP_MEMORY_SCOPE_AGENT) < NWG_) { }
            }
            __syncthreads();
        }

        v8f acc0, acc1;
        #pragma unroll
        for (int v = 0; v < 8; ++v) { acc0[v] = 0.0f; acc1[v] = 0.0f; }

        if (step > 0) {
            const u16* hb = h_ws + (size_t)(dir * 2 + ((step - 1) & 1)) * B_ * H_;
            const u16* ap = hb + (size_t)(m0 + l16) * H_ + lhi * 8;
            #pragma unroll 4
            for (int k0 = 0; k0 < H_; k0 += 32) {
                FragBF a, b0, b1;
                a.q[0]  = *(const uint4*)(ap + k0);
                a.q[1]  = *(const uint4*)(ap + k0 + 16);
                b0.q[0] = *(const uint4*)(bp0 + k0);
                b0.q[1] = *(const uint4*)(bp0 + k0 + 16);
                b1.q[0] = *(const uint4*)(bp1 + k0);
                b1.q[1] = *(const uint4*)(bp1 + k0 + 16);
                acc0 = __builtin_amdgcn_wmma_f32_16x16x32_bf16(
                    false, a.v, false, b0.v, (short)0, acc0, false, false);
                acc1 = __builtin_amdgcn_wmma_f32_16x16x32_bf16(
                    false, a.v, false, b1.v, (short)0, acc1, false, false);
            }
        }

        // add precomputed xp, stage z in LDS (C layout: row=v+8*lhi, col=l16)
        {
            const int colbase = g * 512 + wg * 32;
            #pragma unroll
            for (int v = 0; v < 8; ++v) {
                const int b = m0 + v + lhi * 8;
                const size_t rowoff = ((size_t)b * T_ + t) * G4_;
                zbuf[g][b][l16]      = acc0[v] + xp[rowoff + colbase + l16];
                zbuf[g][b][16 + l16] = acc1[v] + xp[rowoff + colbase + 16 + l16];
            }
            if (step + 1 < T_) {   // warm caches for next timestep's xp slice
                const int tn = dir ? (T_ - 2 - step) : (step + 1);
                __builtin_prefetch(&xp[((size_t)(m0 + lhi * 8) * T_ + tn) * G4_ + colbase + l16], 0, 1);
            }
        }
        __syncthreads();

        u16* hb_out = h_ws + (size_t)(dir * 2 + (step & 1)) * B_ * H_;
        #pragma unroll
        for (int i = 0; i < 4; ++i) {
            const int cell = tid + i * 256;
            const int b = cell >> 5, j = cell & 31;
            const float zi = zbuf[0][b][j];
            const float zf = zbuf[1][b][j];
            const float zo = zbuf[2][b][j];
            const float zg = zbuf[3][b][j];
            const float c  = sigf(zf) * cbuf[b][j] + sigf(zi) * tanhf(zg);
            const float h  = sigf(zo) * tanhf(c);
            cbuf[b][j] = c;
            hb_out[(size_t)b * H_ + wg * 32 + j] = f2bf(h);
            out[((size_t)b * T_ + t) * (2 * H_) + dir * H_ + wg * 32 + j] = h;
        }
        __threadfence();
        __syncthreads();
        if (tid == 0) {
            __hip_atomic_fetch_add(&cnt[step], 1, __ATOMIC_RELEASE,
                                   __HIP_MEMORY_SCOPE_AGENT);
        }
    }
}

// ---------------------------------------------------------------------------
extern "C" void kernel_launch(void* const* d_in, const int* in_sizes, int n_in,
                              void* d_out, int out_size, void* d_ws, size_t ws_size,
                              hipStream_t stream)
{
    const float* x    = (const float*)d_in[0];
    const float* W_fw = (const float*)d_in[1];
    const float* b_fw = (const float*)d_in[2];
    const float* W_bw = (const float*)d_in[3];
    const float* b_bw = (const float*)d_in[4];
    float* out = (float*)d_out;

    char* ws = (char*)d_ws;
    const size_t xp_bytes  = 2ull * M_ * G4_ * sizeof(float);   // 256 MB
    const size_t xbf_bytes = (size_t)B_ * T_ * D_ * sizeof(u16); // 16 MB
    const size_t wT_bytes  = 2ull * G4_ * KW_ * sizeof(u16);     // 8 MB
    const size_t h_bytes   = 2ull * 2 * B_ * H_ * sizeof(u16);   // 128 KB
    const size_t off_xbf = xp_bytes;
    const size_t off_wT  = off_xbf + xbf_bytes;
    const size_t off_h   = off_wT + wT_bytes;
    const size_t off_cnt = off_h + h_bytes;

    float* xp_ws = (float*)(ws);
    u16*   xbf   = (u16*)(ws + off_xbf);
    u16*   wT_ws = (u16*)(ws + off_wT);
    u16*   h_ws  = (u16*)(ws + off_h);
    int*   cnt_ws = (int*)(ws + off_cnt);

    const int NE = B_ * T_ * D_ + 2 * G4_ * KW_;   // 12M conversion elements
    prep_kernel<<<dim3(NE / 256), 256, 0, stream>>>(x, W_fw, W_bw, xbf, wT_ws, cnt_ws);
    xp_gemm_kernel<<<dim3(G4_ / 128, M_ / 128, 2), 256, 0, stream>>>(
        xbf, wT_ws, b_fw, b_bw, xp_ws);
    lstm_scan_kernel<<<dim3(2 * NWG_), 256, 0, stream>>>(
        xp_ws, wT_ws, h_ws, cnt_ws, out);
}
